// TinyAttention_40716289966308
// MI455X (gfx1250) — compile-verified, matrix operand order checked by reading, not run on hardware
//
#include <hip/hip_runtime.h>
#include <hip/hip_bf16.h>

typedef __attribute__((ext_vector_type(16))) __bf16   v16bf;
typedef __attribute__((ext_vector_type(8)))  float    v8f;
typedef __attribute__((ext_vector_type(8)))  unsigned v8u;

#define D_ATTN 64
#define NSEQ   2048
#define DFFN   1536
#define NB     8
#define MTOT   (NB * NSEQ)     // 16384 total rows
#define NQKV   (3 * D_ATTN)    // 192
#define DOUT   (DFFN / 2)      // 768

// ---- helpers -------------------------------------------------------------
__device__ __forceinline__ unsigned short f2bf_u(float f) {
  unsigned u = __builtin_bit_cast(unsigned, f);
  u += 0x7FFFu + ((u >> 16) & 1u);            // round-to-nearest-even
  return (unsigned short)(u >> 16);
}
// packed fp32 pair -> 2 x bf16 in one VALU op (CDNA5 v_cvt_pk_bf16_f32)
__device__ __forceinline__ unsigned cvt_pk_bf16(float a, float b) {
  unsigned r;
  asm("v_cvt_pk_bf16_f32 %0, %1, %2" : "=v"(r) : "v"(a), "v"(b));
  return r;
}
// K-index pattern for 16-bit A/B fragments (ISA 7.12.2): VGPR v, half-wave h
__device__ __forceinline__ int koff(int v, int half) {
  return ((v & 3) << 1) + ((v >> 2) << 4) + (half << 3);
}
// CDNA5 async copy: global -> LDS, 16B per lane, tracked by ASYNCcnt
__device__ __forceinline__ void async_b128(unsigned lds_off, const void* g) {
  asm volatile("global_load_async_to_lds_b128 %0, %1, off"
               :: "v"(lds_off), "v"(g) : "memory");
}
__device__ __forceinline__ void wait_async0() {
  asm volatile("s_wait_asynccnt 0x0" ::: "memory");
}

// ---- kernel 1: qkv = x @ W_qkv + b ; x read ONCE (block owns all 12 N-tiles)
// stores Q,K row-major bf16 and V transposed [b][d][n] bf16
__global__ void __launch_bounds__(256)
qkv_kernel(const float* __restrict__ x, const float* __restrict__ Wqkv,
           const float* __restrict__ bqkv,
           unsigned short* __restrict__ qws, unsigned short* __restrict__ kws,
           unsigned short* __restrict__ vT)
{
  __shared__ unsigned short WldsT[NQKV * 32];   // W chunk, [n][k_local], 12KB

  const int tid  = threadIdx.x;
  const int lane = tid & 31;
  const int wave = tid >> 5;
  const int half = lane >> 4;
  const int l16  = lane & 15;
  const int m0   = blockIdx.x * 128 + wave * 16;
  const int arow = m0 + l16;

  v8f c[12] = {};

  for (int kk = 0; kk < DFFN; kk += 32) {
    __syncthreads();                            // WldsT reuse guard
    // cooperative stage: W[kk..kk+32)[0..192) -> LDS transposed, packed cvt
#pragma unroll
    for (int i = 0; i < 12; ++i) {
      int pid = tid + 256 * i;                  // 0..3071 (pairs along k)
      int n   = pid % NQKV;
      int kl  = (pid / NQKV) * 2;
      float f0 = Wqkv[(size_t)(kk + kl) * NQKV + n];
      float f1 = Wqkv[(size_t)(kk + kl + 1) * NQKV + n];
      *(unsigned*)(&WldsT[n * 32 + kl]) = cvt_pk_bf16(f0, f1);
    }
    __syncthreads();

    // A fragment: this wave's 16 rows of x, converted once per K-step
    v8u au;
#pragma unroll
    for (int v = 0; v < 8; ++v) {
      int ko = koff(v, half);
      float2 ap = *(const float2*)(x + (size_t)arow * DFFN + kk + ko);
      au[v] = cvt_pk_bf16(ap.x, ap.y);
    }
    v16bf a = __builtin_bit_cast(v16bf, au);

#pragma unroll
    for (int nt = 0; nt < 12; ++nt) {
      v8u bu;
#pragma unroll
      for (int v = 0; v < 8; ++v) {
        int ko = koff(v, half);
        bu[v] = *(const unsigned*)(&WldsT[(nt * 16 + l16) * 32 + ko]);
      }
      v16bf b = __builtin_bit_cast(v16bf, bu);
      c[nt] = __builtin_amdgcn_wmma_f32_16x16x32_bf16(false, a, false, b,
                                                      (short)0, c[nt], false, false);
    }
  }

#pragma unroll
  for (int nt = 0; nt < 12; ++nt) {
    const int n = nt * 16 + l16;
    const float bias = bqkv[n];
#pragma unroll
    for (int j = 0; j < 8; ++j) {
      int m = m0 + j + 8 * half;
      unsigned short h = f2bf_u(c[nt][j] + bias);
      if (n < 64) {
        qws[(size_t)m * 64 + n] = h;
      } else if (n < 128) {
        kws[(size_t)m * 64 + (n - 64)] = h;
      } else {
        int bb = m >> 11, key = m & (NSEQ - 1);
        vT[((size_t)bb * 64 + (n - 128)) * NSEQ + key] = h;
      }
    }
  }
}

// ---- kernel 2: flash attention; double-buffered async K/V staging --------
__global__ void __launch_bounds__(256)
attn_kernel(const unsigned short* __restrict__ qws,
            const unsigned short* __restrict__ kws,
            const unsigned short* __restrict__ vT,
            unsigned short* __restrict__ attn_out)
{
  __shared__ unsigned short Kstage[2][32][64];  // 2 x (32 keys x 64 d), 8KB
  __shared__ unsigned short Vstage[2][64][32];  // 2 x (64 d x 32 keys), 8KB
  __shared__ float          Slds[8][16][33];
  __shared__ float          alphaLds[8][16];
  __shared__ float          sumLds[8][16];
  __shared__ unsigned short Plds[8][16][32];

  const int tid  = threadIdx.x;
  const int lane = tid & 31;
  const int wave = tid >> 5;
  const int half = lane >> 4;
  const int l16  = lane & 15;

  const int qt   = blockIdx.x * 8 + wave;       // 16-row query tile id
  const int bb   = qt >> 7;                     // batch; all 8 waves same batch
  const int qrow = qt * 16 + l16;

  const float scale = 0.125f;                   // 1/sqrt(64)

  v8f acc0 = {}, acc1 = {}, acc2 = {}, acc3 = {};
  float m_run = -3.0e38f, s_run = 0.0f;         // row stats in lanes 0..15

  // Q fragments (reused for every key chunk)
  v16bf aq0, aq1;
  {
    v8u u0, u1;
#pragma unroll
    for (int v = 0; v < 8; ++v) {
      int ko = koff(v, half);
      u0[v] = *(const unsigned*)(qws + (size_t)qrow * 64 + ko);
      u1[v] = *(const unsigned*)(qws + (size_t)qrow * 64 + 32 + ko);
    }
    aq0 = __builtin_bit_cast(v16bf, u0);
    aq1 = __builtin_bit_cast(v16bf, u1);
  }

  const size_t kbase = (size_t)bb * NSEQ * 64;
  const size_t vbase = (size_t)bb * 64 * NSEQ;
  // per-lane fixed parts of the staging addresses
  const unsigned short* kgp = kws + kbase + (size_t)(tid >> 3) * 64 + (tid & 7) * 8;
  const unsigned short* vgp = vT + vbase + (size_t)(tid >> 2) * NSEQ + (tid & 3) * 8;
  const unsigned kldsL = (unsigned)(size_t)(&Kstage[0][0][0]) + tid * 16;
  const unsigned vldsL = (unsigned)(size_t)(&Vstage[0][0][0]) + tid * 16;

  // prologue: stage chunk 0 into buffer 0
  async_b128(kldsL, kgp);
  async_b128(vldsL, vgp);

  for (int kb = 0; kb < NSEQ; kb += 32) {
    const int buf = (kb >> 5) & 1;
    wait_async0();
    __syncthreads();                 // staged data visible to every wave
    if (kb + 32 < NSEQ) {            // overlap: stage next chunk into other buf
      async_b128(kldsL + (buf ^ 1) * 4096, kgp + (size_t)(kb + 32) * 64);
      async_b128(vldsL + (buf ^ 1) * 4096, vgp + (kb + 32));
    }

    // ---- energy S = (Q @ K^T) * scale : two 16x16 tiles from LDS ----
#pragma unroll
    for (int t = 0; t < 2; ++t) {
      v8f ce = {};
      v8u bu;
#pragma unroll
      for (int v = 0; v < 8; ++v)
        bu[v] = *(const unsigned*)(&Kstage[buf][16 * t + l16][koff(v, half)]);
      ce = __builtin_amdgcn_wmma_f32_16x16x32_bf16(
          false, aq0, false, __builtin_bit_cast(v16bf, bu), (short)0, ce, false, false);
#pragma unroll
      for (int v = 0; v < 8; ++v)
        bu[v] = *(const unsigned*)(&Kstage[buf][16 * t + l16][32 + koff(v, half)]);
      ce = __builtin_amdgcn_wmma_f32_16x16x32_bf16(
          false, aq1, false, __builtin_bit_cast(v16bf, bu), (short)0, ce, false, false);
#pragma unroll
      for (int j = 0; j < 8; ++j)
        Slds[wave][j + 8 * half][16 * t + l16] = ce[j] * scale;
    }
    __syncthreads();

    // ---- online softmax update (one row per lane 0..15) ----
    if (lane < 16) {
      float cmax = -3.0e38f;
#pragma unroll
      for (int cc = 0; cc < 32; ++cc)
        cmax = fmaxf(cmax, Slds[wave][l16][cc]);
      float mnew  = fmaxf(m_run, cmax);
      float alpha = __expf(m_run - mnew);
      float csum  = 0.0f;
#pragma unroll
      for (int cc = 0; cc < 32; cc += 2) {
        float p0 = __expf(Slds[wave][l16][cc] - mnew);
        float p1 = __expf(Slds[wave][l16][cc + 1] - mnew);
        csum += p0 + p1;
        *(unsigned*)(&Plds[wave][l16][cc]) = cvt_pk_bf16(p0, p1);
      }
      s_run = s_run * alpha + csum;
      m_run = mnew;
      alphaLds[wave][l16] = alpha;
    }
    __syncthreads();

    // ---- rescale accumulators by alpha(row) ----
    float al[8];
#pragma unroll
    for (int j = 0; j < 8; ++j) al[j] = alphaLds[wave][j + 8 * half];
#pragma unroll
    for (int j = 0; j < 8; ++j) {
      acc0[j] *= al[j]; acc1[j] *= al[j]; acc2[j] *= al[j]; acc3[j] *= al[j];
    }

    // ---- P fragment from LDS ----
    v16bf ap;
    {
      v8u pu;
#pragma unroll
      for (int v = 0; v < 8; ++v)
        pu[v] = *(const unsigned*)(&Plds[wave][l16][koff(v, half)]);
      ap = __builtin_bit_cast(v16bf, pu);
    }

    // ---- O += P @ V : four 16x16 output tiles (d = 64), V from LDS ----
#pragma unroll
    for (int t = 0; t < 4; ++t) {
      v8u vu;
#pragma unroll
      for (int v = 0; v < 8; ++v)
        vu[v] = *(const unsigned*)(&Vstage[buf][16 * t + l16][koff(v, half)]);
      v8f& ar = (t == 0 ? acc0 : t == 1 ? acc1 : t == 2 ? acc2 : acc3);
      ar = __builtin_amdgcn_wmma_f32_16x16x32_bf16(
          false, ap, false, __builtin_bit_cast(v16bf, vu), (short)0, ar, false, false);
    }
  }

  if (lane < 16) sumLds[wave][l16] = s_run;
  __syncthreads();

  float inv[8];
#pragma unroll
  for (int j = 0; j < 8; ++j) inv[j] = 1.0f / sumLds[wave][j + 8 * half];

#pragma unroll
  for (int t = 0; t < 4; ++t) {
    const v8f& ar = (t == 0 ? acc0 : t == 1 ? acc1 : t == 2 ? acc2 : acc3);
#pragma unroll
    for (int j = 0; j < 8; ++j) {
      int m = qt * 16 + j + 8 * half;
      attn_out[(size_t)m * 64 + 16 * t + l16] = f2bf_u(ar[j] * inv[j]);
    }
  }
}

// ---- kernel 3: out = attn_out @ W_out + b_out (fp32 result) --------------
__global__ void __launch_bounds__(256)
outproj_kernel(const unsigned short* __restrict__ attn_out,
               const float* __restrict__ Wout, const float* __restrict__ bout,
               float* __restrict__ out)
{
  const int lane = threadIdx.x & 31;
  const int wave = threadIdx.x >> 5;
  const int half = lane >> 4;
  const int l16  = lane & 15;
  const int m0   = blockIdx.x * 128 + wave * 16;
  const int n    = blockIdx.y * 16 + l16;
  const int arow = m0 + l16;

  v8f c = {};
#pragma unroll
  for (int s = 0; s < 2; ++s) {
    v8u au, bu;
#pragma unroll
    for (int v = 0; v < 8; ++v) {
      int ko = koff(v, half);
      au[v] = *(const unsigned*)(attn_out + (size_t)arow * 64 + 32 * s + ko);
      float w0 = Wout[(size_t)(32 * s + ko) * DOUT + n];
      float w1 = Wout[(size_t)(32 * s + ko + 1) * DOUT + n];
      bu[v] = cvt_pk_bf16(w0, w1);
    }
    c = __builtin_amdgcn_wmma_f32_16x16x32_bf16(
        false, __builtin_bit_cast(v16bf, au), false,
        __builtin_bit_cast(v16bf, bu), (short)0, c, false, false);
  }
  const float bias = bout[n];
#pragma unroll
  for (int j = 0; j < 8; ++j)
    out[(size_t)(m0 + j + 8 * half) * DOUT + n] = c[j] + bias;
}

// ---- launch --------------------------------------------------------------
extern "C" void kernel_launch(void* const* d_in, const int* in_sizes, int n_in,
                              void* d_out, int out_size, void* d_ws, size_t ws_size,
                              hipStream_t stream) {
  const float* x    = (const float*)d_in[0];
  const float* Wqkv = (const float*)d_in[1];
  const float* bqkv = (const float*)d_in[2];
  const float* Wout = (const float*)d_in[3];
  const float* bout = (const float*)d_in[4];
  float* out = (float*)d_out;

  unsigned short* qws = (unsigned short*)d_ws;            // [16384][64] bf16
  unsigned short* kws = qws + (size_t)MTOT * 64;          // [16384][64] bf16
  unsigned short* vT  = kws + (size_t)MTOT * 64;          // [8][64][2048] bf16
  unsigned short* att = vT  + (size_t)MTOT * 64;          // [16384][64] bf16

  qkv_kernel<<<dim3(MTOT / 128), 256, 0, stream>>>(x, Wqkv, bqkv, qws, kws, vT);
  attn_kernel<<<dim3(MTOT / 16 / 8), 256, 0, stream>>>(qws, kws, vT, att);
  outproj_kernel<<<dim3(MTOT / 128, DOUT / 16), 256, 0, stream>>>(
      att, Wout, bout, out);
}
